// SelfAttentionBlock_55190329753663
// MI455X (gfx1250) — matrix operand order repaired and checked by reading, compiled, not run on hardware
//
#include <hip/hip_runtime.h>
#include <math.h>

// ---------------------------------------------------------------------------
// Types for gfx1250 WMMA (wave32): v16bf A/B fragments, v8f C/D fragments.
// ---------------------------------------------------------------------------
typedef __bf16 bf16_t;
typedef __attribute__((ext_vector_type(16))) __bf16 v16bf;
typedef __attribute__((ext_vector_type(8)))  __bf16 v8bf;
typedef __attribute__((ext_vector_type(8)))  float  v8f;

union V16U { v16bf v; v8bf h[2]; };

#define WMMA_BF16(A, B, C) \
  __builtin_amdgcn_wmma_f32_16x16x32_bf16(false, (A), false, (B), (short)0, (C), false, false)

// A fragment (16x32 bf16, M x K). Lane L holds row M = L%16; per ISA layout:
//   lanes 0-15 : K = [8g..8g+7] in elems 0-7, K = [16+8g..23+8g] in elems 8-15 (g = L/16)
__device__ __forceinline__ v16bf ld_a_frag(const bf16_t* __restrict__ A, int lda,
                                           int m_base, int k_base, int lane) {
  int m = m_base + (lane & 15);
  int g = (lane >> 4) & 1;
  const bf16_t* p = A + (size_t)m * lda + k_base + 8 * g;
  V16U r;
  r.h[0] = *(const v8bf*)(p);
  r.h[1] = *(const v8bf*)(p + 16);
  return r.v;
}

// B fragment (32x16 bf16, K x N) where B_wmma[k][n] = Bmat[n][k] and Bmat is
// row-major (N x K). Lane L holds column n = n_base + L%16; K = [8g..8g+7] in
// elems 0-7, K = [16+8g..23+8g] in elems 8-15 (g = L/16), matching A.
__device__ __forceinline__ v16bf ld_b_frag(const bf16_t* __restrict__ Bm, int ldb,
                                           int n_base, int k_base, int lane) {
  int n = n_base + (lane & 15);
  int g = (lane >> 4) & 1;
  const bf16_t* p = Bm + (size_t)n * ldb + k_base + 8 * g;
  V16U r;
  r.h[0] = *(const v8bf*)(p);
  r.h[1] = *(const v8bf*)(p + 16);
  return r.v;
}

// ---------------------------------------------------------------------------
// Problem constants (reference shapes are fixed)
// ---------------------------------------------------------------------------
#define NB     4
#define S      1024              // H*W
#define DM     1024              // d_model
#define NH     16                // heads
#define DH     64                // head dim
#define ROWS   (NB * S)          // 4096
#define N_QKV  (3 * DM)          // 3072

// ---------------------------------------------------------------------------
// 1) fp32 -> bf16 weight conversion
// ---------------------------------------------------------------------------
__global__ void k_cvt(const float* __restrict__ src, bf16_t* __restrict__ dst, int n) {
  int i = blockIdx.x * blockDim.x + threadIdx.x;
  if (i < n) dst[i] = (bf16_t)src[i];
}

// ---------------------------------------------------------------------------
// 2) mod = silu(cond) @ ada_w.T + ada_b    (4 x 3072, one wave per output)
// ---------------------------------------------------------------------------
__global__ void k_mod(const float* __restrict__ cond, const float* __restrict__ ada_w,
                      const float* __restrict__ ada_b, float* __restrict__ mod) {
  int wid  = (blockIdx.x * blockDim.x + threadIdx.x) >> 5;
  int lane = threadIdx.x & 31;
  if (wid >= NB * N_QKV) return;
  int b = wid / N_QKV, j = wid % N_QKV;
  const float* c = cond + (size_t)b * DM;
  const float* w = ada_w + (size_t)j * DM;
  float s = 0.f;
  for (int k = lane; k < DM; k += 32) {
    float cv = c[k];
    s += (cv / (1.f + __expf(-cv))) * w[k];
  }
  #pragma unroll
  for (int off = 16; off >= 1; off >>= 1) s += __shfl_xor(s, off, 32);
  if (lane == 0) mod[wid] = s + ada_b[j];
}

// ---------------------------------------------------------------------------
// 3) RMSNorm + adaLN modulation -> bf16 activations (one block per token row)
// ---------------------------------------------------------------------------
__global__ void k_norm_mod(const float* __restrict__ x, const float* __restrict__ norm_scale,
                           const float* __restrict__ mod, bf16_t* __restrict__ xm) {
  int row = blockIdx.x;                   // 0..4095
  int b   = row >> 10;
  const float* xr    = x + (size_t)row * DM;
  const float* shift = mod + (size_t)b * N_QKV;
  const float* scale = shift + DM;
  __shared__ float red[8];
  int t = threadIdx.x, lane = t & 31, wid = t >> 5;
  float v[4], s = 0.f;
  #pragma unroll
  for (int i = 0; i < 4; i++) { v[i] = xr[t + 256 * i]; s += v[i] * v[i]; }
  #pragma unroll
  for (int off = 16; off >= 1; off >>= 1) s += __shfl_xor(s, off, 32);
  if (lane == 0) red[wid] = s;
  __syncthreads();
  if (wid == 0) {
    float r = (lane < 8) ? red[lane] : 0.f;
    #pragma unroll
    for (int off = 4; off >= 1; off >>= 1) r += __shfl_xor(r, off, 8);
    if (lane == 0) red[0] = r;
  }
  __syncthreads();
  float inv = rsqrtf(red[0] * (1.f / (float)DM) + 1e-6f);
  #pragma unroll
  for (int i = 0; i < 4; i++) {
    int c = t + 256 * i;
    float xn = v[i] * norm_scale[c] * inv;
    xm[(size_t)row * DM + c] = (bf16_t)(xn * (1.f + scale[c]) + shift[c]);
  }
}

// ---------------------------------------------------------------------------
// 4) QKV GEMM:  C(4096 x 3072) = A(4096 x 1024) * W^T, bf16 in, bf16 out.
//    Wave computes 32x32 (2x2 WMMA tiles); block = 8 waves -> 64(M) x 128(N).
// ---------------------------------------------------------------------------
__global__ void __launch_bounds__(256)
k_gemm_qkv(const bf16_t* __restrict__ A, const bf16_t* __restrict__ B,
           bf16_t* __restrict__ C) {
  int lane = threadIdx.x & 31;
  int wid  = threadIdx.x >> 5;
  int m0 = blockIdx.y * 64  + (wid & 1) * 32;
  int n0 = blockIdx.x * 128 + (wid >> 1) * 32;
  v8f acc[2][2] = {};
  for (int k = 0; k < DM; k += 32) {
    v16bf a0 = ld_a_frag(A, DM, m0,      k, lane);
    v16bf a1 = ld_a_frag(A, DM, m0 + 16, k, lane);
    v16bf b0 = ld_b_frag(B, DM, n0,      k, lane);
    v16bf b1 = ld_b_frag(B, DM, n0 + 16, k, lane);
    acc[0][0] = WMMA_BF16(a0, b0, acc[0][0]);
    acc[0][1] = WMMA_BF16(a0, b1, acc[0][1]);
    acc[1][0] = WMMA_BF16(a1, b0, acc[1][0]);
    acc[1][1] = WMMA_BF16(a1, b1, acc[1][1]);
  }
  int colg = lane & 15, rowoff = (lane >> 4) * 8;
  #pragma unroll
  for (int tm = 0; tm < 2; tm++)
    #pragma unroll
    for (int tn = 0; tn < 2; tn++) {
      int rb = m0 + tm * 16 + rowoff;
      int cc = n0 + tn * 16 + colg;
      #pragma unroll
      for (int r = 0; r < 8; r++)
        C[(size_t)(rb + r) * N_QKV + cc] = (bf16_t)acc[tm][tn][r];
    }
}

// ---------------------------------------------------------------------------
// 5) q/k cosine-norm (* sqrt(attn_scale)) + 2D RoPE; v transpose.
//    One thread per (b, token, head).
//    Outputs: Q,K as [bh][s][64] bf16;  V as [bh][64][s] bf16 (transposed).
// ---------------------------------------------------------------------------
__global__ void k_qkv_post(const bf16_t* __restrict__ qkv, const float* __restrict__ pos,
                           const float* __restrict__ attn_scale, const float* __restrict__ freqs,
                           bf16_t* __restrict__ Qo, bf16_t* __restrict__ Ko,
                           bf16_t* __restrict__ Vo) {
  int idx = blockIdx.x * blockDim.x + threadIdx.x;
  if (idx >= NB * S * NH) return;
  int h = idx & 15;
  int t = (idx >> 4) & (S - 1);
  int b = idx >> 14;
  size_t row = (size_t)b * S + t;
  const bf16_t* base = qkv + row * N_QKV;
  float p0 = pos[row * 2 + 0], p1 = pos[row * 2 + 1];
  float cs[16], sn[16];
  #pragma unroll
  for (int f = 0; f < 8; f++) {
    float fr = freqs[h * 8 + f];
    float th = p0 * fr, tw = p1 * fr;
    cs[f] = cosf(th); sn[f] = sinf(th);
    cs[f + 8] = cosf(tw); sn[f + 8] = sinf(tw);
  }
  float srt = sqrtf(attn_scale[h]);
  size_t bh = (size_t)(b * NH + h);
  #pragma unroll
  for (int which = 0; which < 2; which++) {          // 0 = q, 1 = k
    const bf16_t* p = base + which * DM + h * DH;
    float v64[DH]; float ss = 0.f;
    #pragma unroll
    for (int d = 0; d < DH; d++) { v64[d] = (float)p[d]; ss += v64[d] * v64[d]; }
    float sc = srt * rsqrtf(ss + 1e-6f);
    #pragma unroll
    for (int d = 0; d < DH; d++) v64[d] *= sc;
    bf16_t* dst = (which == 0 ? Qo : Ko) + (bh * S + t) * DH;
    #pragma unroll
    for (int i = 0; i < 16; i++) {
      float x1 = v64[i], x2 = v64[i + 16];
      dst[i]      = (bf16_t)(x1 * cs[i] - x2 * sn[i]);
      dst[i + 16] = (bf16_t)(x2 * cs[i] + x1 * sn[i]);
    }
    #pragma unroll
    for (int d = 32; d < DH; d++) dst[d] = (bf16_t)v64[d];
  }
  const bf16_t* vp = base + 2 * DM + h * DH;
  #pragma unroll
  for (int d = 0; d < DH; d++)
    Vo[(bh * DH + d) * S + t] = vp[d];
}

// ---------------------------------------------------------------------------
// 6) Flash attention. One wave per 16-query tile; keys in chunks of 32.
//    scores: 4 WMMA (two 16x16 S tiles over K=64)
//    P re-shaped to an A-fragment through a wave-private LDS tile
//    O += P @ V: 4 WMMA (output 16x64 in 4 f32 C tiles)
// ---------------------------------------------------------------------------
__global__ void __launch_bounds__(256)
k_attn(const bf16_t* __restrict__ Q, const bf16_t* __restrict__ Kt,
       const bf16_t* __restrict__ Vt, bf16_t* __restrict__ O) {
  __shared__ bf16_t pbuf[8][16 * 32];
  int lane = threadIdx.x & 31;
  int wid  = threadIdx.x >> 5;
  int qt    = blockIdx.x * 8 + wid;      // global q-tile id: 0..4095
  int bh    = qt >> 6;                   // 0..63
  int qtile = qt & 63;
  const bf16_t* q = Q  + (size_t)bh * S * DH;
  const bf16_t* k = Kt + (size_t)bh * S * DH;
  const bf16_t* v = Vt + (size_t)bh * DH * S;
  int m_base = qtile * 16;

  v16bf qa0 = ld_a_frag(q, DH, m_base, 0,  lane);
  v16bf qa1 = ld_a_frag(q, DH, m_base, 32, lane);

  v8f o_acc[4] = {};
  float m_run[8], l_run[8];
  #pragma unroll
  for (int r = 0; r < 8; r++) { m_run[r] = -1e30f; l_run[r] = 0.f; }
  bf16_t* pb = pbuf[wid];
  int colg = lane & 15, rowoff = (lane >> 4) * 8;

  for (int j = 0; j < S; j += 32) {
    // --- scores for key tiles [j, j+16) and [j+16, j+32)
    v16bf kb0 = ld_b_frag(k, DH, j,      0,  lane);
    v16bf kb1 = ld_b_frag(k, DH, j,      32, lane);
    v16bf kb2 = ld_b_frag(k, DH, j + 16, 0,  lane);
    v16bf kb3 = ld_b_frag(k, DH, j + 16, 32, lane);
    v8f s0 = {}, s1 = {};
    s0 = WMMA_BF16(qa0, kb0, s0);
    s0 = WMMA_BF16(qa1, kb1, s0);
    s1 = WMMA_BF16(qa0, kb2, s1);
    s1 = WMMA_BF16(qa1, kb3, s1);

    // --- online softmax (row reductions across the 16-lane half-groups)
    #pragma unroll
    for (int r = 0; r < 8; r++) {
      float mx = fmaxf(s0[r], s1[r]);
      #pragma unroll
      for (int off = 8; off >= 1; off >>= 1) mx = fmaxf(mx, __shfl_xor(mx, off, 16));
      float m_new = fmaxf(m_run[r], mx);
      float scale = __expf(m_run[r] - m_new);
      float p0 = __expf(s0[r] - m_new);
      float p1 = __expf(s1[r] - m_new);
      float rs = p0 + p1;
      #pragma unroll
      for (int off = 8; off >= 1; off >>= 1) rs += __shfl_xor(rs, off, 16);
      l_run[r] = l_run[r] * scale + rs;
      m_run[r] = m_new;
      s0[r] = p0; s1[r] = p1;
      o_acc[0][r] *= scale; o_acc[1][r] *= scale;
      o_acc[2][r] *= scale; o_acc[3][r] *= scale;
    }

    // --- P (16x32, C layout) -> LDS -> A fragment (wave-private, lockstep)
    #pragma unroll
    for (int r = 0; r < 8; r++) {
      pb[(rowoff + r) * 32 + colg]      = (bf16_t)s0[r];
      pb[(rowoff + r) * 32 + 16 + colg] = (bf16_t)s1[r];
    }
    asm volatile("s_wait_dscnt 0x0" ::: "memory");
    v16bf pa;
    {
      int m = lane & 15, g = lane >> 4;
      const bf16_t* pp = pb + m * 32 + 8 * g;
      V16U r; r.h[0] = *(const v8bf*)pp; r.h[1] = *(const v8bf*)(pp + 16);
      pa = r.v;
    }

    // --- O += P @ V  (V transposed: contiguous over keys per output dim)
    #pragma unroll
    for (int t = 0; t < 4; t++) {
      v16bf vb = ld_b_frag(v, S, t * 16, j, lane);
      o_acc[t] = WMMA_BF16(pa, vb, o_acc[t]);
    }
  }

  // --- finalize: O /= l ; write o as [b][token][h*64 + d] bf16
  int b = bh >> 4, h = bh & 15;
  #pragma unroll
  for (int r = 0; r < 8; r++) {
    float inv = 1.f / l_run[r];
    int tok = m_base + rowoff + r;
    size_t obase = ((size_t)(b * S + tok)) * DM + h * DH;
    O[obase + 0  + colg] = (bf16_t)(o_acc[0][r] * inv);
    O[obase + 16 + colg] = (bf16_t)(o_acc[1][r] * inv);
    O[obase + 32 + colg] = (bf16_t)(o_acc[2][r] * inv);
    O[obase + 48 + colg] = (bf16_t)(o_acc[3][r] * inv);
  }
}

// ---------------------------------------------------------------------------
// 7) Output GEMM + fused residual/gate epilogue (fp32 out).
// ---------------------------------------------------------------------------
__global__ void __launch_bounds__(256)
k_gemm_out(const bf16_t* __restrict__ A, const bf16_t* __restrict__ B,
           const float* __restrict__ x, const float* __restrict__ mod,
           float* __restrict__ out) {
  int lane = threadIdx.x & 31;
  int wid  = threadIdx.x >> 5;
  int m0 = blockIdx.y * 64  + (wid & 1) * 32;
  int n0 = blockIdx.x * 128 + (wid >> 1) * 32;
  v8f acc[2][2] = {};
  for (int k = 0; k < DM; k += 32) {
    v16bf a0 = ld_a_frag(A, DM, m0,      k, lane);
    v16bf a1 = ld_a_frag(A, DM, m0 + 16, k, lane);
    v16bf b0 = ld_b_frag(B, DM, n0,      k, lane);
    v16bf b1 = ld_b_frag(B, DM, n0 + 16, k, lane);
    acc[0][0] = WMMA_BF16(a0, b0, acc[0][0]);
    acc[0][1] = WMMA_BF16(a0, b1, acc[0][1]);
    acc[1][0] = WMMA_BF16(a1, b0, acc[1][0]);
    acc[1][1] = WMMA_BF16(a1, b1, acc[1][1]);
  }
  int colg = lane & 15, rowoff = (lane >> 4) * 8;
  #pragma unroll
  for (int tm = 0; tm < 2; tm++)
    #pragma unroll
    for (int tn = 0; tn < 2; tn++) {
      int rb = m0 + tm * 16 + rowoff;
      int cc = n0 + tn * 16 + colg;
      #pragma unroll
      for (int r = 0; r < 8; r++) {
        int rr = rb + r;
        int bb = rr >> 10;
        float gate = mod[(size_t)bb * N_QKV + 2 * DM + cc];
        size_t o = (size_t)rr * DM + cc;
        out[o] = x[o] + gate * acc[tm][tn][r];
      }
    }
}

// ---------------------------------------------------------------------------
// Host: workspace layout + launch sequence (stream-ordered dependencies)
// ---------------------------------------------------------------------------
extern "C" void kernel_launch(void* const* d_in, const int* in_sizes, int n_in,
                              void* d_out, int out_size, void* d_ws, size_t ws_size,
                              hipStream_t stream) {
  (void)in_sizes; (void)n_in; (void)out_size; (void)ws_size;
  const float* x          = (const float*)d_in[0];
  const float* pos        = (const float*)d_in[1];
  const float* cond       = (const float*)d_in[2];
  const float* norm_scale = (const float*)d_in[3];
  const float* ada_w      = (const float*)d_in[4];
  const float* ada_b      = (const float*)d_in[5];
  const float* qkv_w      = (const float*)d_in[6];
  const float* attn_scale = (const float*)d_in[7];
  const float* out_w      = (const float*)d_in[8];
  const float* freqs      = (const float*)d_in[9];
  float* out = (float*)d_out;

  char* ws = (char*)d_ws;
  size_t off = 0;
  auto alloc = [&](size_t bytes) -> void* {
    off = (off + 255) & ~(size_t)255;
    void* p = ws + off;
    off += bytes;
    return p;
  };
  float*  mod   = (float*) alloc((size_t)NB * N_QKV * 4);
  bf16_t* qkvwb = (bf16_t*)alloc((size_t)N_QKV * DM * 2);
  bf16_t* outwb = (bf16_t*)alloc((size_t)DM * DM * 2);
  bf16_t* xm    = (bf16_t*)alloc((size_t)ROWS * DM * 2);
  bf16_t* qkvb  = (bf16_t*)alloc((size_t)ROWS * N_QKV * 2);
  bf16_t* qn    = (bf16_t*)alloc((size_t)NB * NH * S * DH * 2);
  bf16_t* kn    = (bf16_t*)alloc((size_t)NB * NH * S * DH * 2);
  bf16_t* vt    = (bf16_t*)alloc((size_t)NB * NH * DH * S * 2);
  bf16_t* ob    = (bf16_t*)alloc((size_t)ROWS * DM * 2);

  // 1) weight conversions
  {
    int n1 = N_QKV * DM;
    k_cvt<<<(n1 + 255) / 256, 256, 0, stream>>>(qkv_w, qkvwb, n1);
    int n2 = DM * DM;
    k_cvt<<<(n2 + 255) / 256, 256, 0, stream>>>(out_w, outwb, n2);
  }
  // 2) adaLN modulation vector
  k_mod<<<(NB * N_QKV) / 8, 256, 0, stream>>>(cond, ada_w, ada_b, mod);
  // 3) RMSNorm + modulate -> bf16
  k_norm_mod<<<ROWS, 256, 0, stream>>>(x, norm_scale, mod, xm);
  // 4) QKV GEMM (WMMA)
  k_gemm_qkv<<<dim3(N_QKV / 128, ROWS / 64), 256, 0, stream>>>(xm, qkvwb, qkvb);
  // 5) q/k norm + RoPE + v transpose
  k_qkv_post<<<(NB * S * NH) / 256, 256, 0, stream>>>(qkvb, pos, attn_scale, freqs,
                                                      qn, kn, vt);
  // 6) flash attention (WMMA)
  k_attn<<<(NB * NH * (S / 16)) / 8, 256, 0, stream>>>(qn, kn, vt, ob);
  // 7) output GEMM + residual/gate epilogue (WMMA)
  k_gemm_out<<<dim3(DM / 128, ROWS / 64), 256, 0, stream>>>(ob, outwb, x, mod, out);
}
